// Set2VecReadout_40003325395257
// MI455X (gfx1250) — matrix operand order repaired and checked by reading, not compiled, not run on hardware
//
#include <hip/hip_runtime.h>
#include <math.h>

#define N_NODES      262144
#define DIM          256
#define D_OUT        128
#define NUM_SEG      4096
#define M_BLOCK      128     // rows per block in the MLP kernels (8 waves x 16 rows)
#define K_STEP       32      // WMMA K per step (f16)

typedef _Float16 f16;
typedef __attribute__((ext_vector_type(16))) _Float16 v16h;
typedef __attribute__((ext_vector_type(8)))  _Float16 v8h;
typedef __attribute__((ext_vector_type(8)))  float    v8f;

// ---------------------------------------------------------------------------
// WMMA helpers — layouts per CDNA5 ISA §7.12.2 (wave32)
//   A frag (16x32 f16): lane L: M = L&15, K base = k0 + 8*(L>=16);
//                       halves 0..7 -> K+0..7, halves 8..15 -> K+16..23.
//   B frag (32x16 f16): lane L: N = L&15; lanes 0-15 hold K=0..7,16..23,
//                       lanes 16-31 hold K=8..15,24..31, in half slots 0..15 (same order as A).
//   C/D (16x16 f32):    lane L, vgpr v: M = v + 8*(L>=16), N = L&15.
// ---------------------------------------------------------------------------

__device__ inline v8f wmma_f16(v16h a, v16h b, v8f c) {
    return __builtin_amdgcn_wmma_f32_16x16x32_f16(false, a, false, b, (short)0, c, false, false);
}

__device__ inline float silu(float v) { return v / (1.f + __expf(-v)); }

// A fragment straight from a row-major f32 matrix in global memory (ld = DIM).
// Lanes L and L+16 of one row jointly cover one 128B line per K-step.
__device__ inline v16h load_a_global_f32(const float* __restrict__ base, int row0, int k0, int lane) {
    const int m  = row0 + (lane & 15);
    const int kb = k0 + ((lane >> 4) << 3);
    const float* p = base + (size_t)m * DIM + kb;
    float4 a0 = *(const float4*)(p);
    float4 a1 = *(const float4*)(p + 4);
    float4 a2 = *(const float4*)(p + 16);
    float4 a3 = *(const float4*)(p + 20);
    v16h a;
    a[0]  = (f16)a0.x; a[1]  = (f16)a0.y; a[2]  = (f16)a0.z; a[3]  = (f16)a0.w;
    a[4]  = (f16)a1.x; a[5]  = (f16)a1.y; a[6]  = (f16)a1.z; a[7]  = (f16)a1.w;
    a[8]  = (f16)a2.x; a[9]  = (f16)a2.y; a[10] = (f16)a2.z; a[11] = (f16)a2.w;
    a[12] = (f16)a3.x; a[13] = (f16)a3.y; a[14] = (f16)a3.z; a[15] = (f16)a3.w;
    return a;
}

// A fragment from a row-major f16 LDS tile: two ds_load_b128.
__device__ inline v16h load_a_lds(const f16* __restrict__ p) {
    v8h lo = *(const v8h*)(p);
    v8h hi = *(const v8h*)(p + 16);
    return __builtin_shufflevector(lo, hi, 0,1,2,3,4,5,6,7,8,9,10,11,12,13,14,15);
}

// ---------------------------------------------------------------------------
// One-time weight transform: f32 [K x NOUT] row-major ->
// f16 fragment-order buffer: idx = ((ks*NT + t)*32 + lane)*16 + h,
// with k = ks*32 + 8*(lane>=16) + (h&7) + 16*(h>=8), n = t*16 + (lane&15).
// A B-fragment is then one contiguous 32B chunk per lane.
// ---------------------------------------------------------------------------
__global__ void swizzle_weight_kernel(const float* __restrict__ W, f16* __restrict__ dst,
                                      int K, int NOUT)
{
    const int j = blockIdx.x * blockDim.x + threadIdx.x;
    if (j >= K * NOUT) return;
    const int h    = j & 15;
    const int lane = (j >> 4) & 31;
    const int tf   = j >> 9;
    const int NT   = NOUT >> 4;
    const int t    = tf % NT;
    const int ks   = tf / NT;
    const int k    = ks * K_STEP + ((lane >> 4) << 3) + (h & 7) + ((h >> 3) << 4);
    const int n    = t * 16 + (lane & 15);
    dst[j] = (f16)W[(size_t)k * NOUT + n];
}

// ---------------------------------------------------------------------------
// Kernel A: s = silu(x@W1+b1)@W2+b2, s stored f16.
// 2048 blocks x 256 threads. Each wave owns one 16-row tile x all 16 col tiles.
// Wave-private dataflow: no __syncthreads needed at all.
// ---------------------------------------------------------------------------
__global__ __launch_bounds__(256) void score_mlp_kernel(
    const float* __restrict__ x,
    const f16* __restrict__ w1sw, const float* __restrict__ b1,
    const f16* __restrict__ w2sw, const float* __restrict__ b2,
    f16* __restrict__ s16)
{
    __shared__ f16 hbuf[M_BLOCK * DIM];   // 64 KB, wave-private 16x256 slices

    const int tid  = threadIdx.x;
    const int lane = tid & 31;
    const int wave = tid >> 5;
    const int wrow = blockIdx.x * M_BLOCK + wave * 16;

    v8f zero = {};
    v8f acc[16];
#pragma unroll
    for (int t = 0; t < 16; ++t) acc[t] = zero;

    // ---- layer 1: x @ W1 ----
    const v16h* B1 = (const v16h*)w1sw;
    for (int ks = 0; ks < DIM / K_STEP; ++ks) {
        if (ks < DIM / K_STEP - 1)
            __builtin_prefetch((const void*)(B1 + (ks + 1) * 16 * 32), 0, 0);
        v16h a = load_a_global_f32(x, wrow, ks * K_STEP, lane);
        const v16h* bt = B1 + (size_t)(ks * 16) * 32 + lane;
#pragma unroll
        for (int t = 0; t < 16; ++t)
            acc[t] = wmma_f16(a, bt[t * 32], acc[t]);
    }

    // ---- bias + SiLU -> wave-private LDS h tile ----
    const int mloc = (lane >> 4) << 3;      // C layout row base
    const int ncol = lane & 15;
    {
        f16* hb = hbuf + (wave * 16 + mloc) * DIM + ncol;
#pragma unroll
        for (int t = 0; t < 16; ++t) {
            const float bias = b1[t * 16 + ncol];
#pragma unroll
            for (int v = 0; v < 8; ++v)
                hb[v * DIM + t * 16] = (f16)silu(acc[t][v] + bias);
        }
    }

    // ---- layer 2: h @ W2 ----
#pragma unroll
    for (int t = 0; t < 16; ++t) acc[t] = zero;
    const v16h* B2 = (const v16h*)w2sw;
    const f16* ha = hbuf + (wave * 16 + (lane & 15)) * DIM + ((lane >> 4) << 3);
    for (int ks = 0; ks < DIM / K_STEP; ++ks) {
        if (ks < DIM / K_STEP - 1)
            __builtin_prefetch((const void*)(B2 + (ks + 1) * 16 * 32), 0, 0);
        v16h a = load_a_lds(ha + ks * K_STEP);
        const v16h* bt = B2 + (size_t)(ks * 16) * 32 + lane;
#pragma unroll
        for (int t = 0; t < 16; ++t)
            acc[t] = wmma_f16(a, bt[t * 32], acc[t]);
    }

    // ---- bias + store s as f16 ----
    {
        f16* so = s16 + (size_t)(wrow + mloc) * DIM + ncol;
#pragma unroll
        for (int t = 0; t < 16; ++t) {
            const float bias = b2[t * 16 + ncol];
#pragma unroll
            for (int v = 0; v < 8; ++v)
                so[(size_t)v * DIM + t * 16] = (f16)(acc[t][v] + bias);
        }
    }
}

// ---------------------------------------------------------------------------
// Kernel B: segment start offsets via lower_bound on the sorted index array.
// ---------------------------------------------------------------------------
__global__ void seg_offsets_kernel(const int* __restrict__ index, int* __restrict__ seg_start)
{
    const int b = blockIdx.x * blockDim.x + threadIdx.x;
    if (b > NUM_SEG) return;
    int lo = 0, hi = N_NODES;
    while (lo < hi) {
        int mid = (lo + hi) >> 1;
        if (index[mid] < b) lo = mid + 1; else hi = mid;
    }
    seg_start[b] = lo;
}

// ---------------------------------------------------------------------------
// Kernel C: per-segment online softmax + weighted sum (single streaming pass).
// sx[b,d] = sum_i e_i*x_i / sum_i e_i,  e_i = exp(s_i - max).
// One block per segment, one thread per feature; fully coalesced.
// ---------------------------------------------------------------------------
__global__ __launch_bounds__(256) void seg_softmax_kernel(
    const f16* __restrict__ s16, const float* __restrict__ x,
    const int* __restrict__ seg_start, float* __restrict__ sx)
{
    const int b = blockIdx.x;
    const int d = threadIdx.x;
    const int i0 = seg_start[b];
    const int i1 = seg_start[b + 1];

    float m = -INFINITY, den = 0.f, num = 0.f;
    for (int i = i0; i < i1; ++i) {
        const float sv = (float)s16[(size_t)i * DIM + d];
        const float xv = x[(size_t)i * DIM + d];
        const float mn = fmaxf(m, sv);
        const float scale = __expf(m - mn);   // exp(-inf)=0 on first iteration
        const float e = __expf(sv - mn);
        den = den * scale + e;
        num = num * scale + e * xv;
        m = mn;
    }
    sx[(size_t)b * DIM + d] = (i1 > i0) ? (num / den) : 0.f;
}

// ---------------------------------------------------------------------------
// Kernel D: out = silu(sx@W3+b3)@W4+b4   (256 -> 256 -> 128), 32 blocks.
// ---------------------------------------------------------------------------
__global__ __launch_bounds__(256) void readout_mlp_kernel(
    const float* __restrict__ sx,
    const f16* __restrict__ w3sw, const float* __restrict__ b3,
    const f16* __restrict__ w4sw, const float* __restrict__ b4,
    float* __restrict__ out)
{
    __shared__ f16 hbuf[M_BLOCK * DIM];   // 64 KB

    const int tid  = threadIdx.x;
    const int lane = tid & 31;
    const int wave = tid >> 5;
    const int wrow = blockIdx.x * M_BLOCK + wave * 16;

    v8f zero = {};
    v8f acc[16];
#pragma unroll
    for (int t = 0; t < 16; ++t) acc[t] = zero;

    // ---- layer 1: sx @ W3 (256 -> 256) ----
    const v16h* B3 = (const v16h*)w3sw;
    for (int ks = 0; ks < DIM / K_STEP; ++ks) {
        v16h a = load_a_global_f32(sx, wrow, ks * K_STEP, lane);
        const v16h* bt = B3 + (size_t)(ks * 16) * 32 + lane;
#pragma unroll
        for (int t = 0; t < 16; ++t)
            acc[t] = wmma_f16(a, bt[t * 32], acc[t]);
    }

    const int mloc = (lane >> 4) << 3;
    const int ncol = lane & 15;
    {
        f16* hb = hbuf + (wave * 16 + mloc) * DIM + ncol;
#pragma unroll
        for (int t = 0; t < 16; ++t) {
            const float bias = b3[t * 16 + ncol];
#pragma unroll
            for (int v = 0; v < 8; ++v)
                hb[v * DIM + t * 16] = (f16)silu(acc[t][v] + bias);
        }
    }

    // ---- layer 2: h @ W4 (256 -> 128); 8 col tiles per wave ----
    v8f acc2[8];
#pragma unroll
    for (int t = 0; t < 8; ++t) acc2[t] = zero;
    const v16h* B4 = (const v16h*)w4sw;
    const f16* ha = hbuf + (wave * 16 + (lane & 15)) * DIM + ((lane >> 4) << 3);
    for (int ks = 0; ks < DIM / K_STEP; ++ks) {
        v16h a = load_a_lds(ha + ks * K_STEP);
        const v16h* bt = B4 + (size_t)(ks * 8) * 32 + lane;
#pragma unroll
        for (int t = 0; t < 8; ++t)
            acc2[t] = wmma_f16(a, bt[t * 32], acc2[t]);
    }

    {
        float* oo = out + (size_t)(wrow + mloc) * D_OUT + ncol;
#pragma unroll
        for (int t = 0; t < 8; ++t) {
            const float bias = b4[t * 16 + ncol];
#pragma unroll
            for (int v = 0; v < 8; ++v)
                oo[(size_t)v * D_OUT + t * 16] = acc2[t][v] + bias;
        }
    }
}

// ---------------------------------------------------------------------------
extern "C" void kernel_launch(void* const* d_in, const int* in_sizes, int n_in,
                              void* d_out, int out_size, void* d_ws, size_t ws_size,
                              hipStream_t stream)
{
    const float* x     = (const float*)d_in[0];
    const int*   index = (const int*)  d_in[1];
    const float* W1    = (const float*)d_in[2];
    const float* b1    = (const float*)d_in[3];
    const float* W2    = (const float*)d_in[4];
    const float* b2    = (const float*)d_in[5];
    const float* W3    = (const float*)d_in[6];
    const float* b3    = (const float*)d_in[7];
    const float* W4    = (const float*)d_in[8];
    const float* b4    = (const float*)d_in[9];
    float* out = (float*)d_out;

    // workspace layout (base is 256B aligned; all offsets keep 32B alignment)
    f16*   w1sw  = (f16*)d_ws;                               // 256*256 f16 (128 KB)
    f16*   w2sw  = w1sw + (size_t)DIM * DIM;
    f16*   w3sw  = w2sw + (size_t)DIM * DIM;
    f16*   w4sw  = w3sw + (size_t)DIM * DIM;                 // 256*128 f16 (64 KB)
    float* sx_buf = (float*)(w4sw + (size_t)DIM * D_OUT);    // 4096*256 f32 (4 MB)
    int*   seg    = (int*)(sx_buf + (size_t)NUM_SEG * DIM);  // 4097 int (+pad)
    f16*   s16    = (f16*)(seg + 8192);                      // N*256 f16 (134 MB)

    // one-time weight convert + fragment-order swizzle
    swizzle_weight_kernel<<<(DIM * DIM + 255) / 256, 256, 0, stream>>>(W1, w1sw, DIM, DIM);
    swizzle_weight_kernel<<<(DIM * DIM + 255) / 256, 256, 0, stream>>>(W2, w2sw, DIM, DIM);
    swizzle_weight_kernel<<<(DIM * DIM + 255) / 256, 256, 0, stream>>>(W3, w3sw, DIM, DIM);
    swizzle_weight_kernel<<<(DIM * D_OUT + 255) / 256, 256, 0, stream>>>(W4, w4sw, DIM, D_OUT);

    score_mlp_kernel<<<N_NODES / M_BLOCK, 256, 0, stream>>>(x, w1sw, b1, w2sw, b2, s16);
    seg_offsets_kernel<<<(NUM_SEG + 1 + 255) / 256, 256, 0, stream>>>(index, seg);
    seg_softmax_kernel<<<NUM_SEG, 256, 0, stream>>>(s16, x, seg, sx_buf);
    readout_mlp_kernel<<<NUM_SEG / M_BLOCK, 256, 0, stream>>>(sx_buf, w3sw, b3, w4sw, b4, out);
}
